// GraphSAGE_17575006175717
// MI455X (gfx1250) — matrix operand order repaired and checked by reading, not compiled, not run on hardware
//
#include <hip/hip_runtime.h>

typedef __attribute__((ext_vector_type(16))) __bf16 v16bf;
typedef __attribute__((ext_vector_type(8)))  float  v8f;
typedef __attribute__((ext_vector_type(4)))  unsigned int v4u;
typedef __attribute__((ext_vector_type(8)))  unsigned int v8u;

#define N_NODES 50000
#define N_EDGES 800000
#define D       128
#define NLAYERS 3
#define JKW     (D * NLAYERS)   // 384
#define H2      64
#define NCLS    2
#define EPSBN   1e-5f
#define STRIPS  (N_NODES / 16)  // 3125 exactly
#define SBLK    8               // strips (waves) per block

// ---------------------------------------------------------------------------
// Utility kernels
// ---------------------------------------------------------------------------
__global__ void zero_f32(float* __restrict__ p, int n) {
  int i = blockIdx.x * blockDim.x + threadIdx.x;
  if (i < n) p[i] = 0.0f;
}

__global__ void degree_kernel(const long long* __restrict__ dst, float* __restrict__ deg) {
  int e = blockIdx.x * blockDim.x + threadIdx.x;
  if (e < N_EDGES) atomicAdd(&deg[dst[e]], 1.0f);
}

__global__ void invdeg_kernel(float* __restrict__ deg) {
  int i = blockIdx.x * blockDim.x + threadIdx.x;
  if (i < N_NODES) deg[i] = 1.0f / fmaxf(deg[i], 1.0f);
}

// Convert + transpose weights to bf16 [N][K] row-major (B-fragment friendly).
__global__ void prep_weights(const float* __restrict__ Wl, const float* __restrict__ Wr,
                             const float* __restrict__ Wc1,
                             __bf16* __restrict__ WlT, __bf16* __restrict__ WrT,
                             __bf16* __restrict__ Wc1T) {
  int i = blockIdx.x * blockDim.x + threadIdx.x;
  const int per = NLAYERS * D * D;
  if (i < per) {
    int l = i / (D * D);
    int k = (i / D) % D;
    int n = i % D;
    WlT[l * D * D + n * D + k] = (__bf16)Wl[i];
    WrT[l * D * D + n * D + k] = (__bf16)Wr[i];
  } else if (i < per + JKW * H2) {
    int j = i - per;
    int k = j / H2;
    int n = j % H2;
    Wc1T[n * JKW + k] = (__bf16)Wc1[j];
  }
}

// Fold bias + BN(eval) into per-column scale/shift:  y = t*scale + shift
__global__ void prep_bn(const float* __restrict__ bl, const float* __restrict__ gamma,
                        const float* __restrict__ beta, const float* __restrict__ rmean,
                        const float* __restrict__ rvar,
                        float* __restrict__ scale, float* __restrict__ shift) {
  int i = blockIdx.x * blockDim.x + threadIdx.x;
  if (i < NLAYERS * D) {
    float sc = gamma[i] * rsqrtf(rvar[i] + EPSBN);
    scale[i] = sc;
    shift[i] = beta[i] + (bl[i] - rmean[i]) * sc;
  }
}

// Edge-parallel scatter: one wave per edge, each lane moves 4 floats.
// Working set (~141 MB) fits MI455X's 192 MB L2, so these atomics are L2-resident.
__global__ void scatter_kernel(const float* __restrict__ h, int ldh,
                               const long long* __restrict__ src,
                               const long long* __restrict__ dst,
                               float* __restrict__ agg) {
  int t = blockIdx.x * blockDim.x + threadIdx.x;
  int e = t >> 5;
  int lane = t & 31;
  if (e >= N_EDGES) return;
  long long s = src[e];
  long long d = dst[e];
  const float4 v = *(const float4*)(h + (long)s * ldh + lane * 4);
  float* o = agg + (long)d * D + lane * 4;
  atomicAdd(o + 0, v.x);
  atomicAdd(o + 1, v.y);
  atomicAdd(o + 2, v.z);
  atomicAdd(o + 3, v.w);
}

// ---------------------------------------------------------------------------
// Tensor Data Mover: 1-D contiguous tile, data_size = 4B.  D# per ISA 08 §8.
// Descriptor groups passed as SGPR tuples ("s" ext-vector constraints).
// ---------------------------------------------------------------------------
__device__ __forceinline__ void tdm_load_1d(unsigned lds_off, const void* gp,
                                            unsigned nbytes) {
  unsigned long long ga = (unsigned long long)gp;
  unsigned ndw = nbytes >> 2;  // dwords; must fit 16-bit tile_dim0
  v4u g0 = { 1u,                                     // count=1, user mode
             lds_off,                                // lds_addr
             (unsigned)ga,                           // global_addr[31:0]
             ((unsigned)(ga >> 32) & 0x01FFFFFFu) | 0x80000000u };  // [56:32] | type=2
  v8u g1 = { 0x00020000u,                            // wg_mask=0, data_size=4B
             (ndw & 0xFFFFu) << 16,                  // tensor_dim0[15:0]
             ((ndw >> 16) & 0xFFFFu) | (1u << 16),   // tensor_dim0[31:16], tensor_dim1=1
             (ndw & 0xFFFFu) << 16,                  // tile_dim0
             1u,                                     // tile_dim1=1, tile_dim2=0
             ndw,                                    // tensor_dim0_stride[31:0]
             0u, 0u };                               // stride hi / tensor_dim1_stride
  asm volatile("tensor_load_to_lds %0, %1" :: "s"(g0), "s"(g1) : "memory");
}

// ---------------------------------------------------------------------------
// WMMA fragment helpers (16-bit 16x32 A / 32x16 B, both with contiguous k = 16*(lane/16) + s)
// ---------------------------------------------------------------------------
__device__ __forceinline__ v16bf load_a_f32(const float* __restrict__ base, int ld,
                                            int k0, int lane, float mul) {
  int m = lane & 15;
  int koff = (lane >> 4) * 16;
  const float* p = base + (long)m * ld + k0 + koff;
  v16bf a;
#pragma unroll
  for (int j = 0; j < 8; ++j) {
    a[j]     = (__bf16)(p[j] * mul);
    a[j + 8] = (__bf16)(p[j + 8] * mul);
  }
  return a;
}

__device__ __forceinline__ v16bf load_b_bf16(const __bf16* WT, int ldk,
                                             int n0, int k0, int lane) {
  int n = lane & 15;
  int kb = (lane >> 4) * 16;
  return *(const v16bf*)(WT + (long)(n0 + n) * ldk + k0 + kb);
}

// ---------------------------------------------------------------------------
// Fused SAGE layer:  hout = ReLU( BN( (agg*invdeg)@Wl + bl + h@Wr ) )
// Weights (64 KB) staged once per block into LDS by the Tensor Data Mover;
// each wave computes a 16-row x 128-col strip (8 C tiles, 16 WMMAs per K-step).
// ---------------------------------------------------------------------------
__global__ __launch_bounds__(256) void sage_layer(
    const float* __restrict__ hin, int ldh,
    const float* __restrict__ agg,
    const float* __restrict__ invdeg,
    const __bf16* __restrict__ WlT, const __bf16* __restrict__ WrT,
    const float* __restrict__ scale, const float* __restrict__ shift,
    float* __restrict__ hout, int ldo) {
  __shared__ __attribute__((aligned(64))) __bf16 sW[2 * D * D];  // 64 KB
  int wv = threadIdx.x >> 5;
  int lane = threadIdx.x & 31;
  int strip = blockIdx.x * SBLK + wv;
  bool active = strip < STRIPS;

  // TDM staging: wave0 -> Wl tile, wave1 -> Wr tile (duplicates would be benign).
  if (wv == 0) {
    tdm_load_1d((unsigned)(unsigned long long)&sW[0], WlT, D * D * 2);
  } else if (wv == 1) {
    tdm_load_1d((unsigned)(unsigned long long)&sW[D * D], WrT, D * D * 2);
  }
  __builtin_amdgcn_s_wait_tensorcnt(0);
  __syncthreads();

  if (active) {
    int row0 = strip * 16;
    int m = lane & 15;
    float invd = invdeg[row0 + m];

    v8f acc[8];
#pragma unroll
    for (int nt = 0; nt < 8; ++nt)
#pragma unroll
      for (int j = 0; j < 8; ++j) acc[nt][j] = 0.0f;

#pragma unroll
    for (int kt = 0; kt < 4; ++kt) {
      int k0 = kt * 32;
      v16bf aA = load_a_f32(agg + (long)row0 * D, D, k0, lane, invd);
      v16bf aH = load_a_f32(hin + (long)row0 * ldh, ldh, k0, lane, 1.0f);
#pragma unroll
      for (int nt = 0; nt < 8; ++nt) {
        v16bf bL = load_b_bf16(&sW[0], D, nt * 16, k0, lane);
        acc[nt] = __builtin_amdgcn_wmma_f32_16x16x32_bf16(
            false, aA, false, bL, (short)0, acc[nt], false, false);
        v16bf bR = load_b_bf16(&sW[D * D], D, nt * 16, k0, lane);
        acc[nt] = __builtin_amdgcn_wmma_f32_16x16x32_bf16(
            false, aH, false, bR, (short)0, acc[nt], false, false);
      }
    }

    int rbase = row0 + ((lane >> 4) * 8);
#pragma unroll
    for (int nt = 0; nt < 8; ++nt) {
      int col = nt * 16 + m;
      float sc = scale[col];
      float sh = shift[col];
#pragma unroll
      for (int r = 0; r < 8; ++r) {
        float y = fmaxf(acc[nt][r] * sc + sh, 0.0f);
        hout[(long)(rbase + r) * ldo + col] = y;
      }
    }
  }
}

// ---------------------------------------------------------------------------
// Fused classifier: z = ReLU(jk@Wc1 + bc1) ; logits = z@Wc2 + bc2
// Wc1^T (48 KB) staged via async global->LDS loads (ASYNCcnt); the 64->2 head
// is reduced fully in-register with a width-16 shuffle butterfly.
// ---------------------------------------------------------------------------
__global__ __launch_bounds__(256) void classifier_kernel(
    const float* __restrict__ jk,
    const __bf16* __restrict__ Wc1T,
    const float* __restrict__ bc1,
    const float* __restrict__ Wc2,
    const float* __restrict__ bc2,
    float* __restrict__ out) {
  __shared__ __attribute__((aligned(64))) __bf16 sWc1[H2 * JKW];  // 48 KB
  int wv = threadIdx.x >> 5;
  int lane = threadIdx.x & 31;
  int strip = blockIdx.x * SBLK + wv;
  bool active = strip < STRIPS;
  int m = lane & 15;

  // Cooperative async staging: 3072 x B128, 12 per thread.
  {
    unsigned long long gbase = (unsigned long long)Wc1T;
    unsigned lbase = (unsigned)(unsigned long long)&sWc1[0];
#pragma unroll
    for (int i = 0; i < (H2 * JKW * 2) / (256 * 16); ++i) {
      unsigned idx = i * 256 + threadIdx.x;
      unsigned long long ga = gbase + (unsigned long long)idx * 16;
      unsigned la = lbase + idx * 16;
      asm volatile("global_load_async_to_lds_b128 %0, %1, off"
                   :: "v"(la), "v"(ga) : "memory");
    }
    asm volatile("s_wait_asynccnt 0x0" ::: "memory");
  }
  __syncthreads();

  if (active) {
    int row0 = strip * 16;
    v8f acc[4];
#pragma unroll
    for (int nt = 0; nt < 4; ++nt)
#pragma unroll
      for (int j = 0; j < 8; ++j) acc[nt][j] = 0.0f;

#pragma unroll
    for (int kt = 0; kt < JKW / 32; ++kt) {  // 12 K-steps
      int k0 = kt * 32;
      v16bf aJ = load_a_f32(jk + (long)row0 * JKW, JKW, k0, lane, 1.0f);
#pragma unroll
      for (int nt = 0; nt < 4; ++nt) {
        v16bf b = load_b_bf16(&sWc1[0], JKW, nt * 16, k0, lane);
        acc[nt] = __builtin_amdgcn_wmma_f32_16x16x32_bf16(
            false, aJ, false, b, (short)0, acc[nt], false, false);
      }
    }

    // Per-lane: columns {m, m+16, m+32, m+48}. Head weights for those columns:
    float b1[4], w0[4], w1[4];
#pragma unroll
    for (int nt = 0; nt < 4; ++nt) {
      int col = nt * 16 + m;
      b1[nt] = bc1[col];
      w0[nt] = Wc2[col * NCLS + 0];
      w1[nt] = Wc2[col * NCLS + 1];
    }
    int rbase = (lane >> 4) * 8;  // lanes 16-31 own rows +8
#pragma unroll
    for (int r = 0; r < 8; ++r) {
      float s0 = 0.0f, s1 = 0.0f;
#pragma unroll
      for (int nt = 0; nt < 4; ++nt) {
        float z = fmaxf(acc[nt][r] + b1[nt], 0.0f);
        s0 += z * w0[nt];
        s1 += z * w1[nt];
      }
#pragma unroll
      for (int off = 8; off >= 1; off >>= 1) {  // width-16 butterfly
        s0 += __shfl_xor(s0, off, 16);
        s1 += __shfl_xor(s1, off, 16);
      }
      if (m == 0) {
        long row = (long)(row0 + rbase + r);
        out[row * NCLS + 0] = s0 + bc2[0];
        out[row * NCLS + 1] = s1 + bc2[1];
      }
    }
  }
}

// ---------------------------------------------------------------------------
// Host-side launch
// ---------------------------------------------------------------------------
static inline size_t align256(size_t x) { return (x + 255) & ~size_t(255); }

extern "C" void kernel_launch(void* const* d_in, const int* in_sizes, int n_in,
                              void* d_out, int out_size, void* d_ws, size_t ws_size,
                              hipStream_t stream) {
  const float*     x    = (const float*)d_in[0];
  const long long* ei   = (const long long*)d_in[1];
  const float*     Wl   = (const float*)d_in[2];
  const float*     bl   = (const float*)d_in[3];
  const float*     Wr   = (const float*)d_in[4];
  const float*     gamma= (const float*)d_in[5];
  const float*     beta = (const float*)d_in[6];
  const float*     rmean= (const float*)d_in[7];
  const float*     rvar = (const float*)d_in[8];
  const float*     Wc1  = (const float*)d_in[9];
  const float*     bc1  = (const float*)d_in[10];
  const float*     Wc2  = (const float*)d_in[11];
  const float*     bc2  = (const float*)d_in[12];
  const long long* srcI = ei;
  const long long* dstI = ei + N_EDGES;

  char* w = (char*)d_ws;
  float* invdeg = (float*)w;            w += align256((size_t)N_NODES * 4);
  float* agg    = (float*)w;            w += align256((size_t)N_NODES * D * 4);
  float* jk     = (float*)w;            w += align256((size_t)N_NODES * JKW * 4);
  __bf16* WlT   = (__bf16*)w;           w += align256((size_t)NLAYERS * D * D * 2);
  __bf16* WrT   = (__bf16*)w;           w += align256((size_t)NLAYERS * D * D * 2);
  __bf16* Wc1T  = (__bf16*)w;           w += align256((size_t)H2 * JKW * 2);
  float* bnscale= (float*)w;            w += align256((size_t)NLAYERS * D * 4);
  float* bnshift= (float*)w;            w += align256((size_t)NLAYERS * D * 4);
  (void)ws_size; (void)n_in; (void)in_sizes; (void)out_size;

  {
    int wtot = NLAYERS * D * D + JKW * H2;
    prep_weights<<<(wtot + 255) / 256, 256, 0, stream>>>(Wl, Wr, Wc1, WlT, WrT, Wc1T);
    prep_bn<<<(NLAYERS * D + 255) / 256, 256, 0, stream>>>(bl, gamma, beta, rmean, rvar,
                                                           bnscale, bnshift);
  }
  zero_f32<<<(N_NODES + 255) / 256, 256, 0, stream>>>(invdeg, N_NODES);
  degree_kernel<<<(N_EDGES + 255) / 256, 256, 0, stream>>>(dstI, invdeg);
  invdeg_kernel<<<(N_NODES + 255) / 256, 256, 0, stream>>>(invdeg);

  const float* hin = x;
  int ldh = D;
  const int aggN = N_NODES * D;
  const int nblk = (STRIPS + SBLK - 1) / SBLK;  // 391
  for (int l = 0; l < NLAYERS; ++l) {
    zero_f32<<<(aggN + 255) / 256, 256, 0, stream>>>(agg, aggN);
    scatter_kernel<<<((size_t)N_EDGES * 32 + 255) / 256, 256, 0, stream>>>(
        hin, ldh, srcI, dstI, agg);
    sage_layer<<<nblk, 256, 0, stream>>>(
        hin, ldh, agg, invdeg, WlT + (size_t)l * D * D, WrT + (size_t)l * D * D,
        bnscale + l * D, bnshift + l * D, jk + l * D, JKW);
    hin = jk + l * D;
    ldh = JKW;
  }

  classifier_kernel<<<nblk, 256, 0, stream>>>(jk, Wc1T, bc1, Wc2, bc2, (float*)d_out);
}
